// MarketRegimeMemory_2534030704728
// MI455X (gfx1250) — compile-verified, matrix operand order checked by reading, not run on hardware
//
#include <hip/hip_runtime.h>

// ---------------------------------------------------------------------------
// MarketRegimeMemory forward for MI455X (gfx1250, wave32, WMMA bf16 16x16x32)
// B=32, T=12, N=512, D=512
// ---------------------------------------------------------------------------

#define B_   32
#define T_   12
#define N_   512
#define D_   512
#define TOK  (B_ * N_)     // 16384 tokens
#define ND   (N_ * D_)     // 262144

typedef __bf16 bf16;
typedef __attribute__((ext_vector_type(16))) __bf16 v16bf;
typedef __attribute__((ext_vector_type(8)))  __bf16 v8bf;
typedef __attribute__((ext_vector_type(8)))  float  v8f;

__device__ __forceinline__ float fast_sigmoid(float x) {
  return __builtin_amdgcn_rcpf(1.0f + __expf(-x));
}

// ---------------------------------------------------------------------------
// WMMA helpers.
// A fragment (16x32 bf16, row-major source, ld elements per row):
//   lane<16 : row = lane,     K = {k0..k0+7, k0+16..k0+23}
//   lane>=16: row = lane-16,  K = {k0+8..k0+15, k0+24..k0+31}
// B operand is loaded identically from the *transposed* matrix Bt[n][k].
// C (f32 16x16): VGPR r, lane L -> M = r + 8*(L>=16), N = L & 15.
// ---------------------------------------------------------------------------
__device__ __forceinline__ v16bf load_frag(const bf16* __restrict__ base, int ld,
                                           int row0, int k0) {
  int lane = threadIdx.x & 31;
  int r    = row0 + (lane & 15);
  int ko   = k0 + ((lane >> 4) << 3);
  const bf16* p = base + (size_t)r * ld + ko;
  v8bf lo = *(const v8bf*)p;
  v8bf hi = *(const v8bf*)(p + 16);
  v16bf f;
#pragma unroll
  for (int i = 0; i < 8; ++i) { f[i] = lo[i]; f[i + 8] = hi[i]; }
  return f;
}

__device__ __forceinline__ v8f wmma_bf16(v16bf a, v16bf b, v8f c) {
  return __builtin_amdgcn_wmma_f32_16x16x32_bf16(false, a, false, b,
                                                 (short)0, c, false, false);
}

// One wave computes a 32(row) x 64(col) tile: 2x4 accumulators, K steps of 32.
// Each B fragment is reused for two A fragments -> ~21 FLOP/B from L2.
__device__ __forceinline__ void wave_gemm32(const bf16* __restrict__ A, int lda,
                                            const bf16* __restrict__ Bt, int ldb,
                                            int m0, int n0, int K, v8f acc[2][4]) {
#pragma unroll 2
  for (int kb = 0; kb < K; kb += 32) {
    v16bf a0 = load_frag(A, lda, m0,      kb);
    v16bf a1 = load_frag(A, lda, m0 + 16, kb);
#pragma unroll
    for (int c = 0; c < 4; ++c) {
      v16bf bfg = load_frag(Bt, ldb, n0 + 16 * c, kb);
      acc[0][c] = wmma_bf16(a0, bfg, acc[0][c]);
      acc[1][c] = wmma_bf16(a1, bfg, acc[1][c]);
    }
  }
}

// ---------------------------------------------------------------------------
// K0: weight prep — transpose + fp32->bf16 for q_w, k_w, v_w, rd_w1[:512,:]
//     Wt[n*512+k] = (bf16) W[k*512+n]
// ---------------------------------------------------------------------------
__global__ __launch_bounds__(256)
void k_prep_w(const float* __restrict__ qw, const float* __restrict__ kw,
              const float* __restrict__ vw, const float* __restrict__ rdw1,
              bf16* __restrict__ Wq, bf16* __restrict__ Wk,
              bf16* __restrict__ Wv, bf16* __restrict__ W1t) {
  int idx = blockIdx.x * 256 + threadIdx.x;          // over 512*512
  int n = idx >> 9, k = idx & 511;
  const float* src; bf16* dst;
  switch (blockIdx.y) {
    case 0:  src = qw;   dst = Wq;  break;
    case 1:  src = kw;   dst = Wk;  break;
    case 2:  src = vw;   dst = Wv;  break;
    default: src = rdw1; dst = W1t; break;           // first 512 rows of rd_w1
  }
  dst[idx] = (bf16)src[(size_t)k * 512 + n];
}

// ---------------------------------------------------------------------------
// K1: state = mean of last 3 time steps; fp32 + bf16 copies
// ---------------------------------------------------------------------------
__global__ __launch_bounds__(256)
void k_state(const float* __restrict__ x, float* __restrict__ s32,
             bf16* __restrict__ s16) {
  size_t idx = (size_t)blockIdx.x * 256 + threadIdx.x;   // 0 .. TOK*D-1
  size_t b = idx >> 18, r = idx & (ND - 1);
  const float* xb = x + (((size_t)b * T_ + 9) << 18);
  float v = (xb[r] + xb[r + ND] + xb[r + 2 * ND]) * (1.0f / 3.0f);
  s32[idx] = v;
  s16[idx] = (bf16)v;
}

// ---------------------------------------------------------------------------
// K2: per-(b,d) column mean + unbiased std over N
// ---------------------------------------------------------------------------
__global__ __launch_bounds__(256)
void k_colstats(const float* __restrict__ s32, float* __restrict__ gm,
                float* __restrict__ gs) {
  int b = blockIdx.y;
  int d = blockIdx.x * 256 + threadIdx.x;
  const float* base = s32 + (size_t)b * ND + d;
  float sum = 0.f, sq = 0.f;
  for (int n = 0; n < N_; ++n) {
    float v = base[(size_t)n * D_];
    sum += v; sq += v * v;
  }
  float mean = sum * (1.0f / N_);
  float var  = (sq - (float)N_ * mean * mean) * (1.0f / (N_ - 1));
  gm[b * D_ + d] = mean;
  gs[b * D_ + d] = sqrtf(fmaxf(var, 0.0f));
}

// ---------------------------------------------------------------------------
// K3: gmh[b][j] = rd_b1[j] + sum_i gm[b][i] * rd_w1[512+i][j]
// ---------------------------------------------------------------------------
__global__ __launch_bounds__(256)
void k_gmh(const float* __restrict__ gm, const float* __restrict__ rdw1,
           const float* __restrict__ rdb1, float* __restrict__ gmh) {
  int b = blockIdx.x, tid = threadIdx.x;
  for (int j = tid; j < D_; j += 256) {
    float acc = rdb1[j];
    for (int i = 0; i < D_; ++i)
      acc += gm[b * D_ + i] * rdw1[(size_t)(512 + i) * 512 + j];
    gmh[b * D_ + j] = acc;
  }
}

// ---------------------------------------------------------------------------
// K4: h = silu(state @ rd_w1[:512] + gmh)   — WMMA GEMM [16384,512]x[512,512]
// ---------------------------------------------------------------------------
__global__ __launch_bounds__(128)
void k_regime_h(const bf16* __restrict__ state16, const bf16* __restrict__ w1t,
                const float* __restrict__ gmh, float* __restrict__ h) {
  int wave = threadIdx.x >> 5, lane = threadIdx.x & 31;
  int m0 = blockIdx.y * 128 + wave * 32;
  int n0 = blockIdx.x * 64;
  int b  = m0 >> 9;
  int nlo = lane & 15, mhi = (lane >> 4) << 3;
  v8f acc[2][4];
#pragma unroll
  for (int c = 0; c < 4; ++c) {
    float iv = gmh[b * D_ + n0 + 16 * c + nlo];
#pragma unroll
    for (int r = 0; r < 8; ++r) { acc[0][c][r] = iv; acc[1][c][r] = iv; }
  }
  wave_gemm32(state16, D_, w1t, D_, m0, n0, D_, acc);
#pragma unroll
  for (int rf = 0; rf < 2; ++rf)
#pragma unroll
    for (int c = 0; c < 4; ++c) {
      int nn = n0 + 16 * c + nlo;
#pragma unroll
      for (int r = 0; r < 8; ++r) {
        float xv = acc[rf][c][r];
        h[(size_t)(m0 + rf * 16 + mhi + r) * D_ + nn] = xv * fast_sigmoid(xv);
      }
    }
}

// ---------------------------------------------------------------------------
// K5: regime logits + 3-way softmax (one wave per token)
// ---------------------------------------------------------------------------
__global__ __launch_bounds__(256)
void k_regime_probs(const float* __restrict__ h, const float* __restrict__ w2,
                    const float* __restrict__ b2, float* __restrict__ probs) {
  int wv = threadIdx.x >> 5, lane = threadIdx.x & 31;
  int t = blockIdx.x * 8 + wv;
  const float* hr = h + (size_t)t * D_;
  float a0 = 0.f, a1 = 0.f, a2 = 0.f;
  for (int j = lane; j < D_; j += 32) {
    float hv = hr[j];
    a0 += hv * w2[j * 3 + 0];
    a1 += hv * w2[j * 3 + 1];
    a2 += hv * w2[j * 3 + 2];
  }
#pragma unroll
  for (int off = 16; off > 0; off >>= 1) {
    a0 += __shfl_xor(a0, off, 32);
    a1 += __shfl_xor(a1, off, 32);
    a2 += __shfl_xor(a2, off, 32);
  }
  if (lane == 0) {
    a0 += b2[0]; a1 += b2[1]; a2 += b2[2];
    float mx = fmaxf(a0, fmaxf(a1, a2));
    float e0 = __expf(a0 - mx), e1 = __expf(a1 - mx), e2 = __expf(a2 - mx);
    float inv = 1.0f / (e0 + e1 + e2);
    probs[t * 3 + 0] = e0 * inv;
    probs[t * 3 + 1] = e1 * inv;
    probs[t * 3 + 2] = e2 * inv;
  }
}

// ---------------------------------------------------------------------------
// K6: gate MLP per token: LN(1539) -> 64 (relu) -> 1 (sigmoid) = alpha
// ---------------------------------------------------------------------------
__global__ __launch_bounds__(256)
void k_gate(const float* __restrict__ state, const float* __restrict__ gm,
            const float* __restrict__ gs, const float* __restrict__ probs,
            const float* __restrict__ lnw, const float* __restrict__ lnb,
            const float* __restrict__ w1, const float* __restrict__ b1,
            const float* __restrict__ w2, const float* __restrict__ b2,
            float* __restrict__ alpha) {
  __shared__ float gin[1539];
  __shared__ float red[256];
  __shared__ float rr[64];
  int t = blockIdx.x, b = t >> 9, tid = threadIdx.x;
  const float* sr  = state + (size_t)t * D_;
  const float* gmr = gm + b * D_;
  const float* gsr = gs + b * D_;
  for (int i = tid; i < D_; i += 256) {
    gin[i]        = sr[i];
    gin[512 + i]  = gmr[i];
    gin[1024 + i] = gsr[i];
  }
  if (tid < 3) gin[1536 + tid] = probs[t * 3 + tid];
  __syncthreads();
  float s = 0.f, q = 0.f;
  for (int i = tid; i < 1539; i += 256) { float v = gin[i]; s += v; q += v * v; }
  red[tid] = s; __syncthreads();
  for (int k = 128; k > 0; k >>= 1) { if (tid < k) red[tid] += red[tid + k]; __syncthreads(); }
  float mean = red[0] * (1.0f / 1539.0f); __syncthreads();
  red[tid] = q; __syncthreads();
  for (int k = 128; k > 0; k >>= 1) { if (tid < k) red[tid] += red[tid + k]; __syncthreads(); }
  float var = red[0] * (1.0f / 1539.0f) - mean * mean; __syncthreads();
  float rs = rsqrtf(var + 1e-5f);
  for (int i = tid; i < 1539; i += 256)
    gin[i] = (gin[i] - mean) * rs * lnw[i] + lnb[i];
  __syncthreads();
  int j = tid & 63, part = tid >> 6;
  float acc = 0.f;
  for (int i = part; i < 1539; i += 4) acc += gin[i] * w1[(size_t)i * 64 + j];
  red[tid] = acc; __syncthreads();
  if (tid < 64) {
    float tot = red[tid] + red[tid + 64] + red[tid + 128] + red[tid + 192] + b1[tid];
    rr[tid] = fmaxf(tot, 0.0f);
  }
  __syncthreads();
  if (tid == 0) {
    float l = b2[0];
    for (int jj = 0; jj < 64; ++jj) l += rr[jj] * w2[jj];
    alpha[t] = fast_sigmoid(l);
  }
}

// ---------------------------------------------------------------------------
// K7: QKV GEMM [16384,512]x[512,512] + bias.
//   mode 0: Q  (scale by 1/sqrt(D), row-major bf16)
//   mode 1: K  (row-major bf16 — already the transposed-B layout for QK^T)
//   mode 2: V  (store bf16 transposed per batch: Vt[b][d][n])
// ---------------------------------------------------------------------------
__global__ __launch_bounds__(128)
void k_qkv(const bf16* __restrict__ state16, const bf16* __restrict__ wt,
           const float* __restrict__ bias, bf16* __restrict__ out_rm,
           bf16* __restrict__ out_tr, int mode, float scale) {
  int wave = threadIdx.x >> 5, lane = threadIdx.x & 31;
  int m0 = blockIdx.y * 128 + wave * 32;
  int n0 = blockIdx.x * 64;
  int nlo = lane & 15, mhi = (lane >> 4) << 3;
  v8f acc[2][4];
#pragma unroll
  for (int c = 0; c < 4; ++c) {
    float iv = bias[n0 + 16 * c + nlo];
#pragma unroll
    for (int r = 0; r < 8; ++r) { acc[0][c][r] = iv; acc[1][c][r] = iv; }
  }
  wave_gemm32(state16, D_, wt, D_, m0, n0, D_, acc);
#pragma unroll
  for (int rf = 0; rf < 2; ++rf)
#pragma unroll
    for (int c = 0; c < 4; ++c) {
      int nn = n0 + 16 * c + nlo;
#pragma unroll
      for (int r = 0; r < 8; ++r) {
        float xv = acc[rf][c][r] * scale;
        int m = m0 + rf * 16 + mhi + r;
        if (mode == 2) {
          int b = m >> 9, n = m & 511;
          out_tr[(size_t)b * ND + (size_t)nn * N_ + n] = (bf16)xv;
        } else {
          out_rm[(size_t)m * D_ + nn] = (bf16)xv;
        }
      }
    }
}

// ---------------------------------------------------------------------------
// K8/K10: batched GEMM [512,512]x[512,512]^T-layout per batch, fp32 out
// ---------------------------------------------------------------------------
__global__ __launch_bounds__(128)
void k_bgemm(const bf16* __restrict__ A, const bf16* __restrict__ Bt,
             float* __restrict__ C) {
  int b = blockIdx.z;
  const bf16* Ab = A + (size_t)b * ND;
  const bf16* Bb = Bt + (size_t)b * ND;
  float* Cb = C + (size_t)b * ND;
  int wave = threadIdx.x >> 5, lane = threadIdx.x & 31;
  int m0 = blockIdx.y * 128 + wave * 32;
  int n0 = blockIdx.x * 64;
  int nlo = lane & 15, mhi = (lane >> 4) << 3;
  v8f acc[2][4];
#pragma unroll
  for (int c = 0; c < 4; ++c)
#pragma unroll
    for (int r = 0; r < 8; ++r) { acc[0][c][r] = 0.0f; acc[1][c][r] = 0.0f; }
  wave_gemm32(Ab, D_, Bb, D_, m0, n0, D_, acc);
#pragma unroll
  for (int rf = 0; rf < 2; ++rf)
#pragma unroll
    for (int c = 0; c < 4; ++c) {
      int nn = n0 + 16 * c + nlo;
#pragma unroll
      for (int r = 0; r < 8; ++r)
        Cb[(size_t)(m0 + rf * 16 + mhi + r) * N_ + nn] = acc[rf][c][r];
    }
}

// ---------------------------------------------------------------------------
// K9: row softmax of scores -> attn output; mixed = alpha*I + (1-alpha)*attn
// ---------------------------------------------------------------------------
__global__ __launch_bounds__(256)
void k_softmax_mix(const float* __restrict__ scores, const float* __restrict__ alpha,
                   float* __restrict__ attn_out, bf16* __restrict__ mixed) {
  __shared__ float red[256];
  int t = blockIdx.x, tid = threadIdx.x;
  int n = t & (N_ - 1);
  const float* row = scores + (size_t)t * N_;
  float v0 = row[tid], v1 = row[tid + 256];
  red[tid] = fmaxf(v0, v1); __syncthreads();
  for (int s = 128; s > 0; s >>= 1) { if (tid < s) red[tid] = fmaxf(red[tid], red[tid + s]); __syncthreads(); }
  float mx = red[0]; __syncthreads();
  float e0 = __expf(v0 - mx), e1 = __expf(v1 - mx);
  red[tid] = e0 + e1; __syncthreads();
  for (int s = 128; s > 0; s >>= 1) { if (tid < s) red[tid] += red[tid + s]; __syncthreads(); }
  float inv = 1.0f / red[0];
  float a = alpha[t];
  float at0 = e0 * inv, at1 = e1 * inv;
  attn_out[(size_t)t * N_ + tid]       = at0;
  attn_out[(size_t)t * N_ + tid + 256] = at1;
  float m0v = (1.0f - a) * at0 + ((tid == n) ? a : 0.0f);
  float m1v = (1.0f - a) * at1 + ((tid + 256 == n) ? a : 0.0f);
  mixed[(size_t)t * N_ + tid]       = (bf16)m0v;
  mixed[(size_t)t * N_ + tid + 256] = (bf16)m1v;
}

// ---------------------------------------------------------------------------
// K11: out = LayerNorm(mixed@V + state)
// ---------------------------------------------------------------------------
__global__ __launch_bounds__(256)
void k_out_ln(const float* __restrict__ ao, const float* __restrict__ state,
              const float* __restrict__ lnw, const float* __restrict__ lnb,
              float* __restrict__ out) {
  __shared__ float red[256];
  int t = blockIdx.x, tid = threadIdx.x;
  size_t base = (size_t)t * D_;
  float v0 = ao[base + tid]       + state[base + tid];
  float v1 = ao[base + tid + 256] + state[base + tid + 256];
  red[tid] = v0 + v1; __syncthreads();
  for (int s = 128; s > 0; s >>= 1) { if (tid < s) red[tid] += red[tid + s]; __syncthreads(); }
  float mean = red[0] * (1.0f / D_); __syncthreads();
  red[tid] = v0 * v0 + v1 * v1; __syncthreads();
  for (int s = 128; s > 0; s >>= 1) { if (tid < s) red[tid] += red[tid + s]; __syncthreads(); }
  float var = red[0] * (1.0f / D_) - mean * mean;
  float rs = rsqrtf(var + 1e-5f);
  out[base + tid]       = (v0 - mean) * rs * lnw[tid] + lnb[tid];
  out[base + tid + 256] = (v1 - mean) * rs * lnw[tid + 256] + lnb[tid + 256];
}

// ---------------------------------------------------------------------------
extern "C" void kernel_launch(void* const* d_in, const int* in_sizes, int n_in,
                              void* d_out, int out_size, void* d_ws, size_t ws_size,
                              hipStream_t stream) {
  (void)in_sizes; (void)n_in; (void)out_size; (void)ws_size;
  const float* x      = (const float*)d_in[0];
  const float* rd_w1  = (const float*)d_in[1];
  const float* rd_b1  = (const float*)d_in[2];
  const float* rd_w2  = (const float*)d_in[3];
  const float* rd_b2  = (const float*)d_in[4];
  const float* g_ln_w = (const float*)d_in[5];
  const float* g_ln_b = (const float*)d_in[6];
  const float* g_w1   = (const float*)d_in[7];
  const float* g_b1   = (const float*)d_in[8];
  const float* g_w2   = (const float*)d_in[9];
  const float* g_b2   = (const float*)d_in[10];
  const float* q_w    = (const float*)d_in[11];
  const float* q_b    = (const float*)d_in[12];
  const float* k_w    = (const float*)d_in[13];
  const float* k_b    = (const float*)d_in[14];
  const float* v_w    = (const float*)d_in[15];
  const float* v_b    = (const float*)d_in[16];
  const float* ln_w   = (const float*)d_in[17];
  const float* ln_b   = (const float*)d_in[18];

  // Output tuple flat: out[B,N,D], alpha[B,N,1], attn[B,N,N], probs[B,N,3]
  float* out_y     = (float*)d_out;
  float* out_alpha = out_y + (size_t)TOK * D_;
  float* out_attn  = out_alpha + TOK;
  float* out_probs = out_attn + (size_t)TOK * N_;

  // Workspace map (lifetime-aliased, ~130.2 MB)
  char*  ws      = (char*)d_ws;
  float* state32 = (float*)(ws);                         // 32 MB  (lives to end)
  bf16*  state16 = (bf16*)(ws + (size_t)(32u  << 20));   // 16 MB
  bf16*  Qb      = (bf16*)(ws + (size_t)(48u  << 20));   // 16 MB, later 'mixed'
  bf16*  Kb      = (bf16*)(ws + (size_t)(64u  << 20));   // 16 MB
  bf16*  Vt      = (bf16*)(ws + (size_t)(80u  << 20));   // 16 MB
  float* bigf    = (float*)(ws + (size_t)(96u << 20));   // 32 MB: h / scores / attnout
  bf16*  Wq      = (bf16*)(ws + (size_t)(128u << 20));   // 4 x 0.5 MB
  bf16*  Wk      = Wq  + ND;
  bf16*  Wv      = Wk  + ND;
  bf16*  W1t     = Wv  + ND;
  float* gm      = (float*)(ws + (size_t)(130u << 20));  // 64 KB
  float* gs      = gm + B_ * D_;
  float* gmh     = gs + B_ * D_;
  bf16*  mixed   = Qb;                                   // alias: Q dead after scores

  const float qscale = 0.04419417382415922f;             // 1/sqrt(512)

  k_prep_w<<<dim3(1024, 4), 256, 0, stream>>>(q_w, k_w, v_w, rd_w1, Wq, Wk, Wv, W1t);
  k_state<<<dim3((TOK * D_) / 256), 256, 0, stream>>>(x, state32, state16);
  k_colstats<<<dim3(2, B_), 256, 0, stream>>>(state32, gm, gs);
  k_gmh<<<dim3(B_), 256, 0, stream>>>(gm, rd_w1, rd_b1, gmh);
  k_regime_h<<<dim3(8, TOK / 128), 128, 0, stream>>>(state16, W1t, gmh, bigf);
  k_regime_probs<<<dim3(TOK / 8), 256, 0, stream>>>(bigf, rd_w2, rd_b2, out_probs);
  k_gate<<<dim3(TOK), 256, 0, stream>>>(state32, gm, gs, out_probs,
                                        g_ln_w, g_ln_b, g_w1, g_b1, g_w2, g_b2,
                                        out_alpha);
  k_qkv<<<dim3(8, TOK / 128), 128, 0, stream>>>(state16, Wq, q_b, Qb, nullptr, 0, qscale);
  k_qkv<<<dim3(8, TOK / 128), 128, 0, stream>>>(state16, Wk, k_b, Kb, nullptr, 1, 1.0f);
  k_qkv<<<dim3(8, TOK / 128), 128, 0, stream>>>(state16, Wv, v_b, nullptr, Vt, 2, 1.0f);
  k_bgemm<<<dim3(8, 4, B_), 128, 0, stream>>>(Qb, Kb, bigf);                 // scores
  k_softmax_mix<<<dim3(TOK), 256, 0, stream>>>(bigf, out_alpha, out_attn, mixed);
  k_bgemm<<<dim3(8, 4, B_), 128, 0, stream>>>(mixed, Vt, bigf);              // mixed@V
  k_out_ln<<<dim3(TOK), 256, 0, stream>>>(bigf, state32, ln_w, ln_b, out_y);
}
